// MoE_86182813761870
// MI455X (gfx1250) — compile-verified, matrix operand order checked by reading, not run on hardware
//
#include <hip/hip_runtime.h>

// ---------------------------------------------------------------------------
// MoE (top-2 of 8 experts, SwiGLU) for MI455X / gfx1250, wave32.
// Core math: v_wmma_f32_16x16x32_bf16, double-buffered LDS pipeline,
// GLOBAL_LOAD_ASYNC_TO_LDS for the bf16 activation tiles (probe-confirmed
// builtin; signature: (v4i32 AS1* src, v4i32 AS3* dst, imm offset, imm cpol)).
// ---------------------------------------------------------------------------

#define NTOK 4096   // B*S
#define DDIM 1024
#define HDIM 4096
#define NEXP 8

typedef __bf16 bf16_t;
typedef __bf16 bf16x4  __attribute__((ext_vector_type(4)));
typedef __bf16 bf16x8  __attribute__((ext_vector_type(8)));
typedef __bf16 bf16x16 __attribute__((ext_vector_type(16)));
typedef float  v8f     __attribute__((ext_vector_type(8)));

// LDS row pitch in bf16 elements: 40 (=80 bytes): keeps 16B alignment for
// b128 chunks (80 = 5*16) while spreading rows across the 64 banks.
#define TSTR 40

// ---- gfx1250 async global->LDS path (compile-time probed) -----------------
#if defined(__has_builtin)
#if __has_builtin(__builtin_amdgcn_global_load_async_to_lds_b128)
#define USE_ASYNC_LDS 1
#endif
#endif
#ifndef USE_ASYNC_LDS
#define USE_ASYNC_LDS 0
#endif

#if USE_ASYNC_LDS
typedef int v4i_vs __attribute__((vector_size(4 * sizeof(int))));
typedef __attribute__((address_space(1))) v4i_vs* as1_v4i_ptr;
typedef __attribute__((address_space(3))) v4i_vs* as3_v4i_ptr;

__device__ __forceinline__ void async_copy_b128(const void* g, void* lds) {
    __builtin_amdgcn_global_load_async_to_lds_b128(
        (as1_v4i_ptr)g, (as3_v4i_ptr)lds, 0, 0);
}
__device__ __forceinline__ void async_wait0() {
#if __has_builtin(__builtin_amdgcn_s_wait_asynccnt)
    __builtin_amdgcn_s_wait_asynccnt(0);
#else
    asm volatile("s_wait_asynccnt 0" ::: "memory");
#endif
}
#endif

// Build a 16-elem bf16 WMMA fragment from two 8-elem LDS chunks, matching the
// ISA 16-bit A/B layout: lane half holds K={k0..k0+7} then K={k0+16..k0+23}.
__device__ __forceinline__ bf16x16 frag16(const bf16_t* base) {
    bf16x8 lo = *(const bf16x8*)(base);
    bf16x8 hi = *(const bf16x8*)(base + 16);
    return __builtin_shufflevector(lo, hi, 0, 1, 2, 3, 4, 5, 6, 7,
                                           8, 9, 10, 11, 12, 13, 14, 15);
}

__device__ __forceinline__ bf16x4 cvt4(float4 f) {
    bf16x4 v;
    v[0] = (bf16_t)f.x; v[1] = (bf16_t)f.y;
    v[2] = (bf16_t)f.z; v[3] = (bf16_t)f.w;
    return v;
}

// ---------------------------------------------------------------------------
// Kernel 0: x fp32 -> bf16
// ---------------------------------------------------------------------------
__global__ void moe_cvt_x(const float* __restrict__ x, bf16_t* __restrict__ xb,
                          int n) {
    int i = (blockIdx.x * blockDim.x + threadIdx.x) * 4;
    if (i >= n) return;
    *(bf16x4*)(xb + i) = cvt4(*(const float4*)(x + i));
}

// ---------------------------------------------------------------------------
// Kernel 1: router logits + softmax + top-2 + renormalized combine weights
// ---------------------------------------------------------------------------
__global__ void moe_router(const float* __restrict__ x,
                           const float* __restrict__ Wr,
                           float* __restrict__ logits,
                           float* __restrict__ cw) {
    const int wid  = threadIdx.x >> 5;
    const int lane = threadIdx.x & 31;
    const int tok  = blockIdx.x * 8 + wid;
    const float* xr = x + (size_t)tok * DDIM;

    float acc[NEXP];
#pragma unroll
    for (int e = 0; e < NEXP; ++e) acc[e] = 0.f;
    for (int d = lane; d < DDIM; d += 32) {
        float xv = xr[d];
#pragma unroll
        for (int e = 0; e < NEXP; ++e) acc[e] += xv * Wr[e * DDIM + d];
    }
#pragma unroll
    for (int e = 0; e < NEXP; ++e) {
#pragma unroll
        for (int off = 16; off > 0; off >>= 1)
            acc[e] += __shfl_xor(acc[e], off, 32);
    }
    if (lane == 0) {
        float m = acc[0];
#pragma unroll
        for (int e = 1; e < NEXP; ++e) m = fmaxf(m, acc[e]);
        float p[NEXP], s = 0.f;
#pragma unroll
        for (int e = 0; e < NEXP; ++e) { p[e] = __expf(acc[e] - m); s += p[e]; }
        float inv = 1.f / s;
#pragma unroll
        for (int e = 0; e < NEXP; ++e) p[e] *= inv;
        int i1 = 0;
#pragma unroll
        for (int e = 1; e < NEXP; ++e) if (p[e] > p[i1]) i1 = e;
        int i2 = (i1 == 0) ? 1 : 0;
#pragma unroll
        for (int e = 0; e < NEXP; ++e) if (e != i1 && p[e] > p[i2]) i2 = e;
        float rs = 1.f / (p[i1] + p[i2]);
#pragma unroll
        for (int e = 0; e < NEXP; ++e) {
            logits[tok * NEXP + e] = acc[e];
            cw[tok * NEXP + e] =
                (e == i1) ? p[i1] * rs : ((e == i2) ? p[i2] * rs : 0.f);
        }
    }
}

// ---------------------------------------------------------------------------
// Kernel 2 (per expert): h = silu(x @ Wg^T) * (x @ Wu^T), bf16 out.
// 256 threads = 8 waves; block tile 128 tok x 64 H; wave tile 32x32.
// Double-buffered LDS, one barrier per K-step, 8 WMMA per wave per K-step.
// ---------------------------------------------------------------------------
__global__ __launch_bounds__(256)
void moe_gateup(const bf16_t* __restrict__ xb, const float* __restrict__ Wg,
                const float* __restrict__ Wu, bf16_t* __restrict__ hb) {
    __shared__ bf16_t As[2][128 * TSTR];
    __shared__ bf16_t Bg[2][64 * TSTR];
    __shared__ bf16_t Bu[2][64 * TSTR];

    const int tid   = threadIdx.x;
    const int wid   = tid >> 5;
    const int lane  = tid & 31;
    const int mw    = (wid & 3) * 32;    // wave M offset within tile
    const int nw    = (wid >> 2) * 32;   // wave N offset within tile
    const int mBase = blockIdx.x * 128;
    const int nBase = blockIdx.y * 64;
    const int lrow  = lane & 15;
    const int khalf = (lane >> 4) * 8;

    // staging coordinates: A = 2 b128 chunks/thread, B = 2 float4/thread/mat
    const int ar0 = tid >> 2, ac0 = (tid & 3) * 8;  // rows ar0, ar0+64
    const int br0 = tid >> 3, bc0 = (tid & 7) * 4;  // rows br0, br0+32

    const v8f zero = {};
    v8f cg[2][2], cu[2][2];
#pragma unroll
    for (int mi = 0; mi < 2; ++mi)
#pragma unroll
        for (int ni = 0; ni < 2; ++ni) { cg[mi][ni] = zero; cu[mi][ni] = zero; }

    // ---- prologue: stage K-step 0 into buffer 0 ----
    {
#if USE_ASYNC_LDS
        async_copy_b128(xb + (size_t)(mBase + ar0) * DDIM + ac0,
                        &As[0][ar0 * TSTR + ac0]);
        async_copy_b128(xb + (size_t)(mBase + ar0 + 64) * DDIM + ac0,
                        &As[0][(ar0 + 64) * TSTR + ac0]);
#else
        *(uint4*)&As[0][ar0 * TSTR + ac0] =
            *(const uint4*)(xb + (size_t)(mBase + ar0) * DDIM + ac0);
        *(uint4*)&As[0][(ar0 + 64) * TSTR + ac0] =
            *(const uint4*)(xb + (size_t)(mBase + ar0 + 64) * DDIM + ac0);
#endif
#pragma unroll
        for (int i = 0; i < 2; ++i) {
            int row = br0 + i * 32;
            size_t off = (size_t)(nBase + row) * DDIM + bc0;
            *(bf16x4*)&Bg[0][row * TSTR + bc0] = cvt4(*(const float4*)(Wg + off));
            *(bf16x4*)&Bu[0][row * TSTR + bc0] = cvt4(*(const float4*)(Wu + off));
        }
#if USE_ASYNC_LDS
        async_wait0();
#endif
        __syncthreads();
    }

    const int NK = DDIM / 32;
    for (int kk = 0; kk < NK; ++kk) {
        const int p = kk & 1;
        const bool more = (kk + 1) < NK;
        float4 fg[2], fu[2];
#if !USE_ASYNC_LDS
        uint4 av[2];
#endif
        // 1) issue next-step global traffic early (overlaps with WMMA below)
        if (more) {
            const int k = (kk + 1) * 32;
#if USE_ASYNC_LDS
            async_copy_b128(xb + (size_t)(mBase + ar0) * DDIM + k + ac0,
                            &As[1 - p][ar0 * TSTR + ac0]);
            async_copy_b128(xb + (size_t)(mBase + ar0 + 64) * DDIM + k + ac0,
                            &As[1 - p][(ar0 + 64) * TSTR + ac0]);
#else
            av[0] = *(const uint4*)(xb + (size_t)(mBase + ar0) * DDIM + k + ac0);
            av[1] = *(const uint4*)(xb + (size_t)(mBase + ar0 + 64) * DDIM + k + ac0);
#endif
#pragma unroll
            for (int i = 0; i < 2; ++i) {
                size_t off = (size_t)(nBase + br0 + i * 32) * DDIM + k + bc0;
                fg[i] = *(const float4*)(Wg + off);
                fu[i] = *(const float4*)(Wu + off);
            }
            if (kk + 2 < NK) {  // global_prefetch_b8 two steps ahead
                __builtin_prefetch(Wg + (size_t)(nBase + br0) * DDIM + (kk + 2) * 32 + bc0, 0, 1);
                __builtin_prefetch(Wu + (size_t)(nBase + br0) * DDIM + (kk + 2) * 32 + bc0, 0, 1);
            }
        }
        // 2) load ALL fragments, then one WMMA chain (single dscnt wait)
        bf16x16 af[2], bgf[2], buf_[2];
#pragma unroll
        for (int mi = 0; mi < 2; ++mi)
            af[mi] = frag16(&As[p][(mw + mi * 16 + lrow) * TSTR + khalf]);
#pragma unroll
        for (int ni = 0; ni < 2; ++ni) {
            bgf[ni]  = frag16(&Bg[p][(nw + ni * 16 + lrow) * TSTR + khalf]);
            buf_[ni] = frag16(&Bu[p][(nw + ni * 16 + lrow) * TSTR + khalf]);
        }
#pragma unroll
        for (int mi = 0; mi < 2; ++mi)
#pragma unroll
            for (int ni = 0; ni < 2; ++ni) {
                cg[mi][ni] = __builtin_amdgcn_wmma_f32_16x16x32_bf16(
                    false, af[mi], false, bgf[ni], (short)0, cg[mi][ni], false, false);
                cu[mi][ni] = __builtin_amdgcn_wmma_f32_16x16x32_bf16(
                    false, af[mi], false, buf_[ni], (short)0, cu[mi][ni], false, false);
            }
        // 3) drain staged data into the other buffer
        if (more) {
#pragma unroll
            for (int i = 0; i < 2; ++i) {
                int row = br0 + i * 32;
                *(bf16x4*)&Bg[1 - p][row * TSTR + bc0] = cvt4(fg[i]);
                *(bf16x4*)&Bu[1 - p][row * TSTR + bc0] = cvt4(fu[i]);
            }
#if !USE_ASYNC_LDS
            *(uint4*)&As[1 - p][ar0 * TSTR + ac0] = av[0];
            *(uint4*)&As[1 - p][(ar0 + 64) * TSTR + ac0] = av[1];
#endif
        }
#if USE_ASYNC_LDS
        async_wait0();
#endif
        __syncthreads();
    }

    // epilogue: h = silu(g) * u
    const int hi8 = (lane >> 4) * 8;
#pragma unroll
    for (int mi = 0; mi < 2; ++mi)
#pragma unroll
        for (int r = 0; r < 8; ++r) {
            int grow = mBase + mw + mi * 16 + r + hi8;
#pragma unroll
            for (int ni = 0; ni < 2; ++ni) {
                int gcol = nBase + nw + ni * 16 + lrow;
                float g = cg[mi][ni][r], u = cu[mi][ni][r];
                float h = (g / (1.f + __expf(-g))) * u;
                hb[(size_t)grow * HDIM + gcol] = (bf16_t)h;
            }
        }
}

// ---------------------------------------------------------------------------
// Kernel 3 (per expert): out (+)= (h @ Wd^T) * cw[:, e]
// Same pipeline; block tile 128 tok x 64 D; wave tile 32x32; K over H.
// ---------------------------------------------------------------------------
__global__ __launch_bounds__(256)
void moe_down(const bf16_t* __restrict__ hb, const float* __restrict__ Wd,
              const float* __restrict__ cw, float* __restrict__ out, int e) {
    __shared__ bf16_t As[2][128 * TSTR];
    __shared__ bf16_t Bs[2][64 * TSTR];

    const int tid   = threadIdx.x;
    const int wid   = tid >> 5;
    const int lane  = tid & 31;
    const int mw    = (wid & 3) * 32;
    const int nw    = (wid >> 2) * 32;
    const int mBase = blockIdx.x * 128;
    const int nBase = blockIdx.y * 64;
    const int lrow  = lane & 15;
    const int khalf = (lane >> 4) * 8;

    const int ar0 = tid >> 2, ac0 = (tid & 3) * 8;
    const int br0 = tid >> 3, bc0 = (tid & 7) * 4;

    const v8f zero = {};
    v8f acc[2][2];
#pragma unroll
    for (int mi = 0; mi < 2; ++mi)
#pragma unroll
        for (int ni = 0; ni < 2; ++ni) acc[mi][ni] = zero;

    {   // prologue: K-step 0 -> buffer 0
#if USE_ASYNC_LDS
        async_copy_b128(hb + (size_t)(mBase + ar0) * HDIM + ac0,
                        &As[0][ar0 * TSTR + ac0]);
        async_copy_b128(hb + (size_t)(mBase + ar0 + 64) * HDIM + ac0,
                        &As[0][(ar0 + 64) * TSTR + ac0]);
#else
        *(uint4*)&As[0][ar0 * TSTR + ac0] =
            *(const uint4*)(hb + (size_t)(mBase + ar0) * HDIM + ac0);
        *(uint4*)&As[0][(ar0 + 64) * TSTR + ac0] =
            *(const uint4*)(hb + (size_t)(mBase + ar0 + 64) * HDIM + ac0);
#endif
#pragma unroll
        for (int i = 0; i < 2; ++i) {
            int row = br0 + i * 32;
            *(bf16x4*)&Bs[0][row * TSTR + bc0] =
                cvt4(*(const float4*)(Wd + (size_t)(nBase + row) * HDIM + bc0));
        }
#if USE_ASYNC_LDS
        async_wait0();
#endif
        __syncthreads();
    }

    const int NK = HDIM / 32;
    for (int kk = 0; kk < NK; ++kk) {
        const int p = kk & 1;
        const bool more = (kk + 1) < NK;
        float4 fb[2];
#if !USE_ASYNC_LDS
        uint4 av[2];
#endif
        if (more) {
            const int k = (kk + 1) * 32;
#if USE_ASYNC_LDS
            async_copy_b128(hb + (size_t)(mBase + ar0) * HDIM + k + ac0,
                            &As[1 - p][ar0 * TSTR + ac0]);
            async_copy_b128(hb + (size_t)(mBase + ar0 + 64) * HDIM + k + ac0,
                            &As[1 - p][(ar0 + 64) * TSTR + ac0]);
#else
            av[0] = *(const uint4*)(hb + (size_t)(mBase + ar0) * HDIM + k + ac0);
            av[1] = *(const uint4*)(hb + (size_t)(mBase + ar0 + 64) * HDIM + k + ac0);
#endif
#pragma unroll
            for (int i = 0; i < 2; ++i)
                fb[i] = *(const float4*)(Wd + (size_t)(nBase + br0 + i * 32) * HDIM + k + bc0);
            if (kk + 2 < NK)
                __builtin_prefetch(Wd + (size_t)(nBase + br0) * HDIM + (kk + 2) * 32 + bc0, 0, 1);
        }
        bf16x16 af[2], bf[2];
#pragma unroll
        for (int mi = 0; mi < 2; ++mi)
            af[mi] = frag16(&As[p][(mw + mi * 16 + lrow) * TSTR + khalf]);
#pragma unroll
        for (int ni = 0; ni < 2; ++ni)
            bf[ni] = frag16(&Bs[p][(nw + ni * 16 + lrow) * TSTR + khalf]);
#pragma unroll
        for (int mi = 0; mi < 2; ++mi)
#pragma unroll
            for (int ni = 0; ni < 2; ++ni)
                acc[mi][ni] = __builtin_amdgcn_wmma_f32_16x16x32_bf16(
                    false, af[mi], false, bf[ni], (short)0, acc[mi][ni], false, false);
        if (more) {
#pragma unroll
            for (int i = 0; i < 2; ++i) {
                int row = br0 + i * 32;
                *(bf16x4*)&Bs[1 - p][row * TSTR + bc0] = cvt4(fb[i]);
            }
#if !USE_ASYNC_LDS
            *(uint4*)&As[1 - p][ar0 * TSTR + ac0] = av[0];
            *(uint4*)&As[1 - p][(ar0 + 64) * TSTR + ac0] = av[1];
#endif
        }
#if USE_ASYNC_LDS
        async_wait0();
#endif
        __syncthreads();
    }

    const int hi8 = (lane >> 4) * 8;
#pragma unroll
    for (int mi = 0; mi < 2; ++mi)
#pragma unroll
        for (int r = 0; r < 8; ++r) {
            int grow  = mBase + mw + mi * 16 + r + hi8;
            float cwv = cw[grow * NEXP + e];
#pragma unroll
            for (int ni = 0; ni < 2; ++ni) {
                int gcol = nBase + nw + ni * 16 + lrow;
                float v = acc[mi][ni][r] * cwv;
                float* po = out + (size_t)grow * DDIM + gcol;
                if (e == 0) *po = v;    // initialize (d_out is poisoned)
                else        *po += v;   // accumulate across experts
            }
        }
}

// ---------------------------------------------------------------------------
// Host launcher
// ---------------------------------------------------------------------------
extern "C" void kernel_launch(void* const* d_in, const int* in_sizes, int n_in,
                              void* d_out, int out_size, void* d_ws,
                              size_t ws_size, hipStream_t stream) {
    const float* x  = (const float*)d_in[0];  // [N, D]
    const float* Wr = (const float*)d_in[1];  // [E, D]
    const float* Wg = (const float*)d_in[2];  // [E, H, D]
    const float* Wu = (const float*)d_in[3];  // [E, H, D]
    const float* Wd = (const float*)d_in[4];  // [E, D, H]

    float* out    = (float*)d_out;             // [N, D]
    float* logits = out + (size_t)NTOK * DDIM; // [N, E]

    // workspace: cw [N,E] f32 | x bf16 [N,D] | h bf16 [N,H]  (~40.2 MB)
    float*  cw = (float*)d_ws;
    bf16_t* xb = (bf16_t*)((char*)d_ws + (size_t)NTOK * NEXP * 4);
    bf16_t* hb = (bf16_t*)((char*)d_ws + (size_t)NTOK * NEXP * 4 +
                           (size_t)NTOK * DDIM * 2);

    moe_cvt_x<<<(NTOK * DDIM) / (256 * 4), 256, 0, stream>>>(x, xb, NTOK * DDIM);
    moe_router<<<NTOK / 8, 256, 0, stream>>>(x, Wr, logits, cw);

    for (int e = 0; e < NEXP; ++e) {
        moe_gateup<<<dim3(NTOK / 128, HDIM / 64), 256, 0, stream>>>(
            xb, Wg + (size_t)e * HDIM * DDIM, Wu + (size_t)e * HDIM * DDIM, hb);
        moe_down<<<dim3(NTOK / 128, DDIM / 64), 256, 0, stream>>>(
            hb, Wd + (size_t)e * DDIM * HDIM, cw, out, e);
    }
}
